// ScaledDotProductAttention_39848706572857
// MI455X (gfx1250) — compile-verified, matrix operand order checked by reading, not run on hardware
//
#include <hip/hip_runtime.h>

// ---------------------------------------------------------------------------
// Flash attention forward: B=2, S=2048, H=16, D=64, fp16 in/out (MI455X).
//
//   S^T(16k x 16m) = K_tile x Q^T          v_wmma_f32_16x16x32_f16
//   online softmax in fp32, exp2 domain    (Q pre-scaled by 0.125*log2e)
//   O^T(16d x 16m) += V^T_tile x P^T       v_wmma_f32_16x16x32_f16
//
// Workgroup = 4 waves x 32 queries = 128 queries. K / V^T 32-key tiles are
// staged in LDS by the Tensor Data Mover (tensor_load_to_lds, TENSORcnt),
// double-buffered so the DMA overlaps compute; all 4 waves share each tile.
// ---------------------------------------------------------------------------

typedef _Float16 half_t;
typedef __attribute__((ext_vector_type(16))) _Float16 v16h;
typedef __attribute__((ext_vector_type(8)))  _Float16 v8h;
typedef __attribute__((ext_vector_type(8)))  float    v8f;
typedef __attribute__((ext_vector_type(4)))  unsigned v4u;
typedef __attribute__((ext_vector_type(8)))  int      v8i;
typedef __attribute__((ext_vector_type(4)))  int      v4i;

#define B_ 2
#define S_ 2048
#define H_ 16
#define D_ 64
constexpr int HD = H_ * D_;                  // 1024: row stride of [B,S,H,D]

// padded LDS row lengths (halves): K rows 72 (36 dwords), V^T rows 40 (20 dwords)
// -> lane strides 36/20 dwords: all 16 row-lanes hit distinct bank groups.
constexpr int KPAD = 72;
constexpr int VPAD = 40;

#if __has_builtin(__builtin_amdgcn_tensor_load_to_lds)
#define USE_TDM 1
#if __has_include(<hip/amd_detail/amd_gfx1250_TDM.h>)
#define TDM_6ARG 1
#endif
#endif

__device__ __forceinline__ void wait_tensorcnt0() {
#if __has_builtin(__builtin_amdgcn_s_wait_tensorcnt)
  __builtin_amdgcn_s_wait_tensorcnt((short)0);
#else
  asm volatile("s_wait_tensorcnt 0x0" ::: "memory");
#endif
}

#ifdef USE_TDM
// Issue one 2D TDM tile load (global -> LDS) per CDNA5 D# spec (ISA ch.8).
// w1..w5 are pre-shifted descriptor words of group1.
__device__ __forceinline__ void tdm_load_2d(unsigned lds_addr, const void* gaddr,
                                            unsigned w0, unsigned w1, unsigned w2,
                                            unsigned w3, unsigned w4, unsigned w5) {
  const unsigned long long ga = (unsigned long long)(uintptr_t)gaddr;
  // group0: count=1 | lds_addr | global_addr[56:0] | type=2 (bits 127:126)
  v4u g0 = {1u, lds_addr, (unsigned)ga, (unsigned)(ga >> 32) | 0x80000000u};
  v8i g1 = {(int)w0, (int)w1, (int)w2, (int)w3, (int)w4, (int)w5, 0, 0};
  v4i z4 = {0, 0, 0, 0};
#ifdef TDM_6ARG
  v8i z8 = {0, 0, 0, 0, 0, 0, 0, 0};
  __builtin_amdgcn_tensor_load_to_lds(g0, g1, z4, z4, z8, 0);
#else
  __builtin_amdgcn_tensor_load_to_lds(g0, g1, z4, z4, 0);
#endif
}
#endif

// Fragment gather: halves 0..7 at p, halves 8..15 at p+16 (b128 pair).
__device__ __forceinline__ v16h load_frag(const half_t* __restrict__ p) {
  v8h lo = *(const v8h*)(p);
  v8h hi = *(const v8h*)(p + 16);
  v16h r;
#pragma unroll
  for (int i = 0; i < 8; ++i) { r[i] = lo[i]; r[i + 8] = hi[i]; }
  return r;
}

// ---------------------------------------------------------------------------
// Pre-pass: V [B,S,H,D] -> V^T [B*H, D, S] in workspace (8 MiB), LDS-tiled.
// ---------------------------------------------------------------------------
__global__ void __launch_bounds__(256)
vt_transpose_kernel(const half_t* __restrict__ V, half_t* __restrict__ VT) {
  __shared__ half_t tile[64][72];
  const int blk   = blockIdx.x;                // bh*32 + stile
  const int bh    = blk >> 5;
  const int stile = blk & 31;
  const int b = bh >> 4, h = bh & 15;
  const int tid = threadIdx.x;
  const int row = tid >> 2;                    // 0..63
  const int seg = (tid & 3) * 16;

  const half_t* src =
      V + ((size_t)(b * S_ + stile * 64 + row) * H_ + h) * D_ + seg;
#pragma unroll
  for (int j = 0; j < 16; j += 8) {
    v8h v = *(const v8h*)(src + j);
#pragma unroll
    for (int i = 0; i < 8; ++i) tile[row][seg + j + i] = v[i];
  }
  __syncthreads();
  half_t* dst = VT + ((size_t)bh * D_ + row) * S_ + stile * 64 + seg;
#pragma unroll
  for (int j = 0; j < 16; j += 8) {
    v8h v;
#pragma unroll
    for (int i = 0; i < 8; ++i) v[i] = tile[seg + j + i][row];
    *(v8h*)(dst + j) = v;
  }
}

// ---------------------------------------------------------------------------
// Main kernel: grid = B*H*(S/128), block = 128 (4 waves x 32 queries).
// ---------------------------------------------------------------------------
__global__ void __launch_bounds__(128)
fa_fwd_kernel(const half_t* __restrict__ Q,
              const half_t* __restrict__ K,
              const half_t* __restrict__ VT,
              half_t* __restrict__ O) {
  __shared__ __align__(16) half_t ldsK[2][32][KPAD];   // [buf][key][d]
  __shared__ __align__(16) half_t ldsV[2][64][VPAD];   // [buf][d][k]

  const int blk  = blockIdx.x;                 // bh*16 + qblk
  const int bh   = blk >> 4;
  const int qblk = blk & 15;
  const int b = bh >> 4, h = bh & 15;
  const int wave = threadIdx.x >> 5;
  const int lane = threadIdx.x & 31;
  const int lr = lane & 15;                    // fragment row this lane owns
  const int hs = (lane >> 4) * 8;              // 8-wide K-group select

  const int q0 = qblk * 128 + wave * 32;       // first of 32 queries for wave

  // Q fragments for both 16-row tiles, pre-scaled by 1/sqrt(D) * log2(e)
  // so softmax can run in the exp2 domain with bare v_exp_f32.
  const half_t SQ = (half_t)0.18033688f;       // 0.125 * 1.4426950408889634
  v16h qf[2][2];
#pragma unroll
  for (int mt = 0; mt < 2; ++mt) {
    const half_t* qrow =
        Q + ((size_t)(b * S_ + q0 + mt * 16 + lr) * H_ + h) * D_ + hs;
#pragma unroll
    for (int c = 0; c < 2; ++c) {
      v16h f = load_frag(qrow + c * 32);
#pragma unroll
      for (int i = 0; i < 16; ++i) f[i] *= SQ;
      qf[mt][c] = f;
    }
  }

  const half_t* kbase  = K  + ((size_t)b * S_ * H_ + h) * D_;   // + key*HD
  const half_t* vtbase = VT + ((size_t)bh * D_) * S_;           // + d*S + k

  float m_run[2] = {-__builtin_inff(), -__builtin_inff()};
  float l_run[2] = {0.0f, 0.0f};
  v8f ot[2][4];                                // [mtile][dtile], lane = query
#pragma unroll
  for (int mt = 0; mt < 2; ++mt)
#pragma unroll
    for (int dt = 0; dt < 4; ++dt) ot[mt][dt] = (v8f){};

  constexpr int NKB = S_ / 32;                 // 64 key blocks

#ifdef USE_TDM
  // D# group1 words (pre-shifted). data_size=2B (code 1 @bit16), pad_enable
  // @bit20, pad_interval @bits24:22, pad_amount @bits31:25.
  const unsigned KW0 = (1u << 16) | (1u << 20) | (4u << 22) | (3u << 25); // 32DW rows +4DW pad
  const unsigned VW0 = (1u << 16) | (1u << 20) | (3u << 22) | (3u << 25); // 16DW rows +4DW pad
  if (wave == 0) {   // prologue: stage key-block 0 into buffer 0
    tdm_load_2d((unsigned)(uintptr_t)&ldsK[0][0][0], kbase,
                KW0, 64u << 16, 32u << 16, 64u << 16, 32u, (unsigned)HD);
    tdm_load_2d((unsigned)(uintptr_t)&ldsV[0][0][0], vtbase,
                VW0, 32u << 16, 64u << 16, 32u << 16, 64u, (unsigned)S_);
  }
#endif

  for (int kb = 0; kb < NKB; ++kb) {
    const int buf = kb & 1;
#ifdef USE_TDM
    if (wave == 0) wait_tensorcnt0();          // tiles for kb have landed
    __syncthreads();
    if (wave == 0 && kb + 1 < NKB) {           // prefetch next block via TDM
      const int nb = (kb + 1) & 1;
      tdm_load_2d((unsigned)(uintptr_t)&ldsK[nb][0][0],
                  kbase + (size_t)((kb + 1) * 32) * HD,
                  KW0, 64u << 16, 32u << 16, 64u << 16, 32u, (unsigned)HD);
      tdm_load_2d((unsigned)(uintptr_t)&ldsV[nb][0][0],
                  vtbase + (kb + 1) * 32,
                  VW0, 32u << 16, 64u << 16, 32u << 16, 64u, (unsigned)S_);
    }
#else
    // Fallback: cooperative global->LDS copy (b128 in, b128 out).
    __syncthreads();
    {
      const int tid = threadIdx.x;
      const int kr = tid >> 2, kseg = (tid & 3) * 16;      // K: 32 rows x 64
      const half_t* ks = kbase + (size_t)(kb * 32 + kr) * HD + kseg;
      *(v8h*)&ldsK[buf][kr][kseg]     = *(const v8h*)(ks);
      *(v8h*)&ldsK[buf][kr][kseg + 8] = *(const v8h*)(ks + 8);
      const int vr = tid >> 1, vseg = (tid & 1) * 16;      // V^T: 64 rows x 32
      const half_t* vs = vtbase + (size_t)vr * S_ + kb * 32 + vseg;
      *(v8h*)&ldsV[buf][vr][vseg]     = *(const v8h*)(vs);
      *(v8h*)&ldsV[buf][vr][vseg + 8] = *(const v8h*)(vs + 8);
    }
    __syncthreads();
#endif

    // ---- fragments from LDS (ds_load_b128 pairs) ----
    v16h kf[2][2];
#pragma unroll
    for (int tt = 0; tt < 2; ++tt)
#pragma unroll
      for (int c = 0; c < 2; ++c)
        kf[tt][c] = load_frag(&ldsK[buf][tt * 16 + lr][c * 32 + hs]);
    v16h vf[4];
#pragma unroll
    for (int dt = 0; dt < 4; ++dt)
      vf[dt] = load_frag(&ldsV[buf][dt * 16 + lr][hs]);

#pragma unroll
    for (int mt = 0; mt < 2; ++mt) {
      // S^T tiles (fp32), already in exp2 domain via Q pre-scale
      v8f st0 = (v8f){}, st1 = (v8f){};
      st0 = __builtin_amdgcn_wmma_f32_16x16x32_f16(false, kf[0][0], false, qf[mt][0],
                                                   (short)0, st0, false, false);
      st0 = __builtin_amdgcn_wmma_f32_16x16x32_f16(false, kf[0][1], false, qf[mt][1],
                                                   (short)0, st0, false, false);
      st1 = __builtin_amdgcn_wmma_f32_16x16x32_f16(false, kf[1][0], false, qf[mt][0],
                                                   (short)0, st1, false, false);
      st1 = __builtin_amdgcn_wmma_f32_16x16x32_f16(false, kf[1][1], false, qf[mt][1],
                                                   (short)0, st1, false, false);

      float tmax = -__builtin_inff();
#pragma unroll
      for (int i = 0; i < 8; ++i)
        tmax = fmaxf(tmax, fmaxf(st0[i], st1[i]));
      tmax = fmaxf(tmax, __shfl_xor(tmax, 16, 32));
      const float mnew = fmaxf(m_run[mt], tmax);

      v16h pf;
      float rsum = 0.0f;
#pragma unroll
      for (int i = 0; i < 8; ++i) {
        const float p0 = __builtin_amdgcn_exp2f(st0[i] - mnew);
        const float p1 = __builtin_amdgcn_exp2f(st1[i] - mnew);
        rsum += p0 + p1;
        pf[i]     = (half_t)p0;
        pf[i + 8] = (half_t)p1;
      }
      rsum += __shfl_xor(rsum, 16, 32);
      const float corr = __builtin_amdgcn_exp2f(m_run[mt] - mnew);
      l_run[mt] = l_run[mt] * corr + rsum;
      m_run[mt] = mnew;
#pragma unroll
      for (int dt = 0; dt < 4; ++dt)
#pragma unroll
        for (int i = 0; i < 8; ++i) ot[mt][dt][i] *= corr;

#pragma unroll
      for (int dt = 0; dt < 4; ++dt)
        ot[mt][dt] = __builtin_amdgcn_wmma_f32_16x16x32_f16(false, vf[dt], false, pf,
                                                            (short)0, ot[mt][dt],
                                                            false, false);
    }
  }

  // ---- epilogue: O[b, q, h, d] = O^T / l ; per-lane scalar scale ----
#pragma unroll
  for (int mt = 0; mt < 2; ++mt) {
    const float inv = 1.0f / l_run[mt];
    half_t* orow =
        O + ((size_t)(b * S_ + q0 + mt * 16 + lr) * H_ + h) * D_ + hs;
#pragma unroll
    for (int dt = 0; dt < 4; ++dt) {
      v8h o;
#pragma unroll
      for (int i = 0; i < 8; ++i) o[i] = (half_t)(ot[mt][dt][i] * inv);
      *(v8h*)(orow + dt * 16) = o;
    }
  }
}

// ---------------------------------------------------------------------------
extern "C" void kernel_launch(void* const* d_in, const int* in_sizes, int n_in,
                              void* d_out, int out_size, void* d_ws, size_t ws_size,
                              hipStream_t stream) {
  const half_t* Q = (const half_t*)d_in[0];
  const half_t* K = (const half_t*)d_in[1];
  const half_t* V = (const half_t*)d_in[2];
  half_t* Out = (half_t*)d_out;
  half_t* VT  = (half_t*)d_ws;   // needs B*H*D*S*2 = 8 MiB workspace

  (void)in_sizes; (void)n_in; (void)out_size; (void)ws_size;

  vt_transpose_kernel<<<B_ * H_ * (S_ / 64), 256, 0, stream>>>(V, VT);
  fa_fwd_kernel<<<B_ * H_ * (S_ / 128), 128, 0, stream>>>(Q, K, VT, Out);
}
